// Model_51144470560940
// MI455X (gfx1250) — matrix operand order repaired and checked by reading, not compiled, not run on hardware
//
#include <hip/hip_runtime.h>
#include <math.h>

#define B_  32
#define L_  336
#define C_  321
#define P_  96
#define E_  4
#define H_  2048
#define PATCH_ 12
#define R_  (B_*C_)            // 10272 rows = 321 * 32 exactly
#define MT_ 32                 // rows per MoE block (2 x 16-row subtiles)
#define KPAD_ 352              // 336 padded to 11*32
#define INVF 0.9999950000374997f   // 1/sqrt(1+1e-5)

typedef __bf16 bf16;
typedef bf16  v16bf __attribute__((ext_vector_type(16)));
typedef float v8f   __attribute__((ext_vector_type(8)));

// exact GELU (cold paths: mixer, scan)
static __device__ __forceinline__ float gelu_f(float x) {
    return 0.5f * x * (1.0f + erff(x * 0.70710678118654752440f));
}

// branch-free GELU for the hot MoE loop: hardware v_tanh_f32 (CDNA5 TRANS op).
// trailing v_nop satisfies the 1-op TRANS hazard window (compiler can't see asm deps).
static __device__ __forceinline__ float gelu_fast(float x) {
    float x2 = x * x;
    float inner = 0.7978845608028654f * x * __builtin_fmaf(0.044715f, x2, 1.0f);
    float t;
    asm volatile("v_tanh_f32 %0, %1\n\tv_nop" : "=v"(t) : "v"(inner));
    return 0.5f * x * (1.0f + t);
}

// ---------------- stage 0: BN over transposed input --------------------------
__global__ void bn_transpose_kernel(const float* __restrict__ x,
                                    const float* __restrict__ w,
                                    const float* __restrict__ b,
                                    float* __restrict__ h) {
    int idx = blockIdx.x * 256 + threadIdx.x;
    if (idx >= B_ * C_ * L_) return;
    int bb = idx / (C_ * L_);
    int r  = idx % (C_ * L_);
    int c  = r / L_;
    int l  = r % L_;
    h[idx] = x[((size_t)bb * L_ + l) * C_ + c] * (w[r] * INVF) + b[r];
}

__global__ void bn_plain_kernel(const float* __restrict__ in,
                                const float* __restrict__ w,
                                const float* __restrict__ b,
                                float* __restrict__ out) {
    int idx = blockIdx.x * 256 + threadIdx.x;
    if (idx >= B_ * C_ * L_) return;
    int r = idx % (C_ * L_);
    out[idx] = in[idx] * (w[r] * INVF) + b[r];
}

// ---------------- downsample means (k = 8,4,2) -------------------------------
__global__ void downsample_kernel(const float* __restrict__ h,
                                  float* __restrict__ s0,   // (R,42)
                                  float* __restrict__ s1,   // (R,84)
                                  float* __restrict__ s2) { // (R,168)
    int idx = blockIdx.x * 256 + threadIdx.x;
    if (idx >= R_ * 42) return;
    int row = idx / 42, i = idx % 42;
    const float* hp = h + (size_t)row * L_ + i * 8;
    float v0 = hp[0], v1 = hp[1], v2 = hp[2], v3 = hp[3];
    float v4 = hp[4], v5 = hp[5], v6 = hp[6], v7 = hp[7];
    float p01 = v0 + v1, p23 = v2 + v3, p45 = v4 + v5, p67 = v6 + v7;
    s2[(size_t)row * 168 + 4 * i + 0] = p01 * 0.5f;
    s2[(size_t)row * 168 + 4 * i + 1] = p23 * 0.5f;
    s2[(size_t)row * 168 + 4 * i + 2] = p45 * 0.5f;
    s2[(size_t)row * 168 + 4 * i + 3] = p67 * 0.5f;
    s1[(size_t)row * 84 + 2 * i + 0] = (p01 + p23) * 0.25f;
    s1[(size_t)row * 84 + 2 * i + 1] = (p45 + p67) * 0.25f;
    s0[(size_t)row * 42 + i] = (p01 + p23 + p45 + p67) * 0.125f;
}

// ---------------- small dense row-GEMM (mixer MLP stages) --------------------
__global__ void linrow_kernel(const float* __restrict__ in,
                              const float* __restrict__ w,
                              const float* __restrict__ bias,
                              const float* __restrict__ res,
                              float* __restrict__ out,
                              int rows, int din, int dout, int doGelu) {
    int idx = blockIdx.x * 256 + threadIdx.x;
    if (idx >= rows * dout) return;
    int row = idx / dout, j = idx % dout;
    const float* ip = in + (size_t)row * din;
    const float* wp = w  + (size_t)j   * din;
    float acc = bias[j];
    for (int k = 0; k < din; ++k) acc += ip[k] * wp[k];
    if (doGelu) acc = gelu_f(acc);
    if (res)    acc += res[idx];
    out[idx] = acc;
}

// ---------------- DDI sequential scan ---------------------------------------
__global__ void ddi_scan_kernel(const float* __restrict__ dbuf,
                                const float* __restrict__ n1w,
                                const float* __restrict__ n1b,
                                const float* __restrict__ aggw,
                                const float* __restrict__ aggb,
                                float* __restrict__ ddi) {
    __shared__ float sW[144];
    __shared__ float sB[12];
    for (int i = threadIdx.x; i < 144; i += blockDim.x) sW[i] = aggw[i];
    for (int i = threadIdx.x; i < 12;  i += blockDim.x) sB[i] = aggb[i];
    __syncthreads();
    int row = blockIdx.x * blockDim.x + threadIdx.x;
    if (row >= R_) return;
    int c = row % C_;
    const float* dp = dbuf + (size_t)row * L_;
    float* op       = ddi  + (size_t)row * L_;
    float prev[12], nw[12], nb[12];
    #pragma unroll
    for (int i = 0; i < 12; ++i) { prev[i] = dp[i]; op[i] = prev[i]; }
    #pragma unroll
    for (int i = 0; i < 12; ++i) {
        nw[i] = n1w[c * 12 + i] * INVF;
        nb[i] = n1b[c * 12 + i];
    }
    for (int p = 1; p < 28; ++p) {
        float inp[12];
        #pragma unroll
        for (int i = 0; i < 12; ++i) inp[i] = prev[i] * nw[i] + nb[i];
        #pragma unroll
        for (int i = 0; i < 12; ++i) {
            float a = sB[i];
            #pragma unroll
            for (int j = 0; j < 12; ++j) a += inp[j] * sW[i * 12 + j];
            a = gelu_f(a) + dp[p * 12 + i];
            prev[i] = a;
            op[p * 12 + i] = a;
        }
    }
}

// ---------------- gating ------------------------------------------------------
__global__ void gate_kernel(const float* __restrict__ te,
                            const float* __restrict__ gw,
                            const float* __restrict__ gb,
                            float* __restrict__ gates) {
    int row = blockIdx.x * blockDim.x + threadIdx.x;
    if (row >= R_) return;
    const float* tp = te + (size_t)row * L_;
    float lg[4];
    #pragma unroll
    for (int e = 0; e < 4; ++e) {
        float a = gb[e];
        const float* wp = gw + (size_t)e * L_;
        for (int l = 0; l < L_; ++l) a += tp[l] * wp[l];
        lg[e] = a;
    }
    float srt[4] = {lg[0], lg[1], lg[2], lg[3]};
    #define CSWP(i,j) { if (srt[i] > srt[j]) { float t0 = srt[i]; srt[i] = srt[j]; srt[j] = t0; } }
    CSWP(0,1) CSWP(2,3) CSWP(0,2) CSWP(1,3) CSWP(1,2)
    #undef CSWP
    float kth = srt[2];                       // index E-TOPK of ascending sort
    float m = fmaxf(fmaxf(lg[0], lg[1]), fmaxf(lg[2], lg[3]));
    float s[4], sum = 0.f;
    #pragma unroll
    for (int e = 0; e < 4; ++e) { s[e] = expf(lg[e] - m); sum += s[e]; }
    float dec[4];
    #pragma unroll
    for (int e = 0; e < 4; ++e) {
        float se = s[e] / sum;
        dec[e] = (lg[e] < kth) ? 10.0f * logf(se + 1.0f)
                               : 10.0f * (expf(se) - 1.0f);
    }
    float dm = fmaxf(fmaxf(dec[0], dec[1]), fmaxf(dec[2], dec[3]));
    float g2[4], sum2 = 0.f;
    #pragma unroll
    for (int e = 0; e < 4; ++e) { g2[e] = expf(dec[e] - dm); sum2 += g2[e]; }
    #pragma unroll
    for (int e = 0; e < 4; ++e) gates[(size_t)row * 4 + e] = g2[e] / sum2;
}

// ------------- pre-swizzle expert weights into WMMA B fragments --------------
// B 32x16 bf16 layout: lane = {khalf = lane>>4, n = lane&15};
// per lane 16 consecutive K values starting at 32*s + 16*khalf.
__global__ void swz1_kernel(const float* __restrict__ ew1, bf16* __restrict__ b1f) {
    int idx = blockIdx.x * 256 + threadIdx.x;          // (e,t,s,lane)
    if (idx >= E_ * 128 * 11 * 32) return;
    int lane = idx & 31;
    int s    = (idx >> 5) % 11;
    int t    = (idx / (32 * 11)) % 128;
    int e    =  idx / (32 * 11 * 128);
    int n = lane & 15, kh = lane >> 4;
    int kbase = 32 * s + 16 * kh;
    const float* src = ew1 + (size_t)(e * H_ + t * 16 + n) * L_;
    bf16* dst = b1f + (size_t)idx * 16;
    #pragma unroll
    for (int i = 0; i < 16; ++i) {
        int kk = kbase + i;
        dst[i] = (bf16)((kk < L_) ? src[kk] : 0.f);
    }
}

__global__ void swz2_kernel(const float* __restrict__ ew2, bf16* __restrict__ b2f) {
    int idx = blockIdx.x * 256 + threadIdx.x;          // (e,pt,ks,lane)
    if (idx >= E_ * 6 * 64 * 32) return;
    int lane = idx & 31;
    int ks   = (idx >> 5) % 64;
    int pt   = (idx / (32 * 64)) % 6;
    int e    =  idx / (32 * 64 * 6);
    int n = lane & 15, kh = lane >> 4;
    int kbase = 32 * ks + 16 * kh;
    const float* src = ew2 + (size_t)(e * P_ + pt * 16 + n) * H_;
    bf16* dst = b2f + (size_t)idx * 16;
    #pragma unroll
    for (int i = 0; i < 16; ++i) dst[i] = (bf16)src[kbase + i];
}

// ---------------- fused MoE expert kernel (WMMA) -----------------------------
// block = 128 threads = 4 waves; wave e handles expert e for one 32-row tile
// (two 16-row WMMA subtiles -> each B fragment feeds two WMMAs).
__global__ __launch_bounds__(128) void moe_kernel(
        const float* __restrict__ ddi,    // (R, 336)
        const float* __restrict__ gates,  // (R, 4)
        const bf16*  __restrict__ b1f,
        const bf16*  __restrict__ b2f,
        const float* __restrict__ eb1,    // (4, 2048)
        const float* __restrict__ eb2,    // (4, 96)
        float* __restrict__ out) {        // (B, P, C) final output
    __shared__ __align__(16) bf16  smemA[MT_ * KPAD_];      // 32 rows x 352 bf16
    __shared__ __align__(16) bf16  smemH[4][2][16 * 32];    // per-wave hid chunks
    __shared__ float sOut[MT_ * P_];                        // 32 x 96 f32 reduction tile

    const int r0  = blockIdx.x * MT_;
    const int tid = threadIdx.x;

    for (int i = tid; i < MT_ * P_; i += 128) sOut[i] = 0.f;
    // cooperative load of 32 A rows -> bf16, zero-padded K to 352
    for (int idx = tid; idx < MT_ * KPAD_; idx += 128) {
        int row = idx / KPAD_, kk = idx % KPAD_;
        float v = (kk < L_) ? ddi[(size_t)(r0 + row) * L_ + kk] : 0.f;
        smemA[idx] = (bf16)v;
    }
    __syncthreads();

    const int e    = tid >> 5;     // expert = wave id
    const int lane = tid & 31;
    const int n    = lane & 15;
    const int kh   = lane >> 4;

    v8f eo[2][6];
    #pragma unroll
    for (int rt = 0; rt < 2; ++rt)
        #pragma unroll
        for (int pt = 0; pt < 6; ++pt)
            eo[rt][pt] = v8f{0.f,0.f,0.f,0.f,0.f,0.f,0.f,0.f};

    for (int hb = 0; hb < H_; hb += 32) {
        // ---- layer 1: two 16-h tiles x two 16-row subtiles, K = 352 ----
        #pragma unroll
        for (int sub = 0; sub < 2; ++sub) {
            v8f acc0 = v8f{0.f,0.f,0.f,0.f,0.f,0.f,0.f,0.f};
            v8f acc1 = v8f{0.f,0.f,0.f,0.f,0.f,0.f,0.f,0.f};
            int t = (hb >> 4) + sub;
            const bf16* bbase = b1f + ((size_t)(e * 128 + t) * 11) * 512;
            for (int s = 0; s < 11; ++s) {
                v16bf a0, a1, bfrag;
                const bf16* bp = bbase + (size_t)s * 512 + lane * 16;
                *(uint4*)&bfrag       = *(const uint4*)bp;
                *((uint4*)&bfrag + 1) = *(const uint4*)(bp + 8);
                const bf16* ap0 = &smemA[n * KPAD_ + 32 * s + 16 * kh];
                const bf16* ap1 = ap0 + 16 * KPAD_;
                *(uint4*)&a0       = *(const uint4*)ap0;
                *((uint4*)&a0 + 1) = *(const uint4*)(ap0 + 8);
                *(uint4*)&a1       = *(const uint4*)ap1;
                *((uint4*)&a1 + 1) = *(const uint4*)(ap1 + 8);
                acc0 = __builtin_amdgcn_wmma_f32_16x16x32_bf16(
                        false, a0, false, bfrag, (short)0, acc0, false, false);
                acc1 = __builtin_amdgcn_wmma_f32_16x16x32_bf16(
                        false, a1, false, bfrag, (short)0, acc1, false, false);
            }
            // bias + fast GELU (v_tanh_f32), stash to LDS for layer-2 A frags
            int h = hb + 16 * sub + n;
            float bias = eb1[(size_t)e * H_ + h];
            #pragma unroll
            for (int r = 0; r < 8; ++r) {
                int row = r + 8 * kh;
                smemH[e][0][row * 32 + 16 * sub + n] = (bf16)gelu_fast(acc0[r] + bias);
                smemH[e][1][row * 32 + 16 * sub + n] = (bf16)gelu_fast(acc1[r] + bias);
            }
        }
        // ---- layer 2: one K=32 step over hid; B2 frag reused across subtiles ----
        v16bf a20, a21;
        const bf16* hp0 = &smemH[e][0][n * 32 + 16 * kh];
        const bf16* hp1 = &smemH[e][1][n * 32 + 16 * kh];
        *(uint4*)&a20       = *(const uint4*)hp0;
        *((uint4*)&a20 + 1) = *(const uint4*)(hp0 + 8);
        *(uint4*)&a21       = *(const uint4*)hp1;
        *((uint4*)&a21 + 1) = *(const uint4*)(hp1 + 8);
        int ks = hb >> 5;
        #pragma unroll
        for (int pt = 0; pt < 6; ++pt) {
            v16bf b2v;
            const bf16* bp2 = b2f + (size_t)(e * 6 + pt) * 32768 + (size_t)ks * 512 + lane * 16;
            *(uint4*)&b2v       = *(const uint4*)bp2;
            *((uint4*)&b2v + 1) = *(const uint4*)(bp2 + 8);
            eo[0][pt] = __builtin_amdgcn_wmma_f32_16x16x32_bf16(
                        false, a20, false, b2v, (short)0, eo[0][pt], false, false);
            eo[1][pt] = __builtin_amdgcn_wmma_f32_16x16x32_bf16(
                        false, a21, false, b2v, (short)0, eo[1][pt], false, false);
        }
    }

    // gated reduction across the 4 expert waves in LDS (ds_add_f32)
    #pragma unroll
    for (int rt = 0; rt < 2; ++rt)
        #pragma unroll
        for (int pt = 0; pt < 6; ++pt) {
            int p = pt * 16 + n;
            float bias2 = eb2[(size_t)e * P_ + p];
            #pragma unroll
            for (int r = 0; r < 8; ++r) {
                int lrow = rt * 16 + r + 8 * kh;
                float g = gates[(size_t)(r0 + lrow) * 4 + e];
                atomicAdd(&sOut[lrow * P_ + p], g * (eo[rt][pt][r] + bias2));
            }
        }
    __syncthreads();

    // fused transposed store: (row, p) -> out[b, p, c] with row = b*C + c
    for (int i = tid; i < MT_ * P_; i += 128) {
        int lrow = i / P_, p = i % P_;
        int row = r0 + lrow;
        int bb = row / C_, c = row % C_;
        out[((size_t)bb * P_ + p) * C_ + c] = sOut[i];
    }
}

// =============================================================================
extern "C" void kernel_launch(void* const* d_in, const int* in_sizes, int n_in,
                              void* d_out, int out_size, void* d_ws, size_t ws_size,
                              hipStream_t stream) {
    (void)in_sizes; (void)n_in; (void)out_size; (void)ws_size;
    const float* x      = (const float*)d_in[0];
    // d_in[1] = k (unused scalar)
    const float* mdm_w  = (const float*)d_in[2];
    const float* mdm_b  = (const float*)d_in[3];
    const float* m0w1 = (const float*)d_in[4],  *m0b1 = (const float*)d_in[5];
    const float* m0w2 = (const float*)d_in[6],  *m0b2 = (const float*)d_in[7];
    const float* m1w1 = (const float*)d_in[8],  *m1b1 = (const float*)d_in[9];
    const float* m1w2 = (const float*)d_in[10], *m1b2 = (const float*)d_in[11];
    const float* m2w1 = (const float*)d_in[12], *m2b1 = (const float*)d_in[13];
    const float* m2w2 = (const float*)d_in[14], *m2b2 = (const float*)d_in[15];
    const float* dbnw = (const float*)d_in[16], *dbnb = (const float*)d_in[17];
    const float* n1w  = (const float*)d_in[18], *n1b  = (const float*)d_in[19];
    const float* aggw = (const float*)d_in[20], *aggb = (const float*)d_in[21];
    const float* gatw = (const float*)d_in[22], *gatb = (const float*)d_in[23];
    const float* ew1  = (const float*)d_in[24], *eb1  = (const float*)d_in[25];
    const float* ew2  = (const float*)d_in[26], *eb2  = (const float*)d_in[27];
    float* out = (float*)d_out;

    char* ws = (char*)d_ws;
    size_t off = 0;
    auto alloc = [&](size_t bytes) -> void* {
        void* p = ws + off;
        off = (off + bytes + 255) & ~(size_t)255;
        return p;
    };
    float* h     = (float*)alloc((size_t)R_ * L_ * 4);
    float* s0    = (float*)alloc((size_t)R_ * 42 * 4);
    float* s1    = (float*)alloc((size_t)R_ * 84 * 4);
    float* s2    = (float*)alloc((size_t)R_ * 168 * 4);
    float* g     = (float*)alloc((size_t)R_ * 168 * 4);   // reused hidden buffer
    float* cur1  = (float*)alloc((size_t)R_ * 84 * 4);
    float* cur2  = (float*)alloc((size_t)R_ * 168 * 4);
    float* te    = (float*)alloc((size_t)R_ * L_ * 4);
    float* dbuf  = (float*)alloc((size_t)R_ * L_ * 4);
    float* ddi   = (float*)alloc((size_t)R_ * L_ * 4);
    float* gates = (float*)alloc((size_t)R_ * 4 * 4);
    bf16*  b1f   = (bf16*)alloc((size_t)E_ * 128 * 11 * 32 * 16 * 2);
    bf16*  b2f   = (bf16*)alloc((size_t)E_ * 6 * 64 * 32 * 16 * 2);

    auto cdiv = [](int a, int b) { return (a + b - 1) / b; };

    bn_transpose_kernel<<<cdiv(B_*C_*L_,256),256,0,stream>>>(x, mdm_w, mdm_b, h);
    downsample_kernel<<<cdiv(R_*42,256),256,0,stream>>>(h, s0, s1, s2);

    // mixer stage 0: 42 -> 42 -> 84  (+ s1)
    linrow_kernel<<<cdiv(R_*42,256),256,0,stream>>>(s0,   m0w1, m0b1, nullptr, g,    R_, 42, 42, 1);
    linrow_kernel<<<cdiv(R_*84,256),256,0,stream>>>(g,    m0w2, m0b2, s1,      cur1, R_, 42, 84, 0);
    // mixer stage 1: 84 -> 84 -> 168 (+ s2)
    linrow_kernel<<<cdiv(R_*84,256),256,0,stream>>>(cur1, m1w1, m1b1, nullptr, g,    R_, 84, 84, 1);
    linrow_kernel<<<cdiv(R_*168,256),256,0,stream>>>(g,   m1w2, m1b2, s2,      cur2, R_, 84, 168, 0);
    // mixer stage 2: 168 -> 168 -> 336 (+ h) = te
    linrow_kernel<<<cdiv(R_*168,256),256,0,stream>>>(cur2, m2w1, m2b1, nullptr, g,   R_, 168, 168, 1);
    linrow_kernel<<<cdiv(R_*336,256),256,0,stream>>>(g,   m2w2, m2b2, h,       te,   R_, 168, 336, 0);

    bn_plain_kernel<<<cdiv(B_*C_*L_,256),256,0,stream>>>(te, dbnw, dbnb, dbuf);
    ddi_scan_kernel<<<cdiv(R_,128),128,0,stream>>>(dbuf, n1w, n1b, aggw, aggb, ddi);
    gate_kernel<<<cdiv(R_,128),128,0,stream>>>(te, gatw, gatb, gates);

    swz1_kernel<<<cdiv(E_*128*11*32,256),256,0,stream>>>(ew1, b1f);
    swz2_kernel<<<cdiv(E_*6*64*32,256),256,0,stream>>>(ew2, b2f);

    moe_kernel<<<R_/MT_, 128, 0, stream>>>(ddi, gates, b1f, b2f, eb1, eb2, out);
}
